// MultilayerGRU_86681029968322
// MI455X (gfx1250) — matrix operand order repaired and checked by reading, not compiled, not run on hardware
//
#include <hip/hip_runtime.h>
#include <hip/hip_bf16.h>
#include <cstddef>

// ---------------------------------------------------------------------------
// Multilayer GRU (B=32, S=512, I=H=O=1024, L=2) for gfx1250 (MI455X, wave32).
//   * bf16 weights/activations (halves L2 traffic), f32 WMMA accumulate
//   * layer-0 x-projections precomputed in bulk; y-projection in bulk at end
//   * scan kernels: each wave computes full M=32 (two 16x16 tiles) per weight
//     fragment -> weights read exactly once per step; K split 4-way with LDS
//     reduction to shorten the latency-critical WMMA chains
// ---------------------------------------------------------------------------

constexpr int kB = 32, kS = 512, kI = 1024, kH = 1024, kO = 1024, kL = 2;

typedef __attribute__((ext_vector_type(16))) __bf16 bf16x16;
typedef __attribute__((ext_vector_type(8)))  float  floatx8;

union BV16 { bf16x16 v; uint4 q[2]; };

__device__ __forceinline__ unsigned short f2bf(float f) {
  unsigned u = __float_as_uint(f);
  unsigned r = (u + 0x7FFFu + ((u >> 16) & 1u)) >> 16;   // round-to-nearest-even
  return (unsigned short)r;
}
__device__ __forceinline__ float bf2f(unsigned short s) {
  return __uint_as_float(((unsigned)s) << 16);
}

// Lane layouts per CDNA5 ISA 7.12.2 (bf16 16x16x32):
//   A: lanes 0-15 row M=lane, v0..3=K 0..7, v4..7=K 16..23; lanes 16-31 same
//      rows, K 8..15 / 24..31.
//   B: lane holds column N=lane&15; same K split as A (lanes 0-15 K 0..7/16..23, lanes 16-31 K 8..15/24..31).

// acc over K range [k0,k1): two M-tiles (M=32) sharing one B fragment.
__device__ __forceinline__ void wmma_m2(floatx8& c0, floatx8& c1,
    const unsigned short* __restrict__ A, size_t lda,
    const unsigned short* __restrict__ W, size_t ldw,
    int k0, int k1, int lane)
{
  const unsigned short* pa = A + (size_t)(lane & 15) * lda + ((lane >> 4) << 3);
  const unsigned short* pw = W + (size_t)(lane & 15) * ldw + ((lane >> 4) << 3);
  const size_t a1 = (size_t)16 * lda;
  for (int k = k0; k < k1; k += 32) {
    BV16 a0, a1f, b;
    a0.q[0]  = *(const uint4*)(pa + k);
    a0.q[1]  = *(const uint4*)(pa + k + 16);
    a1f.q[0] = *(const uint4*)(pa + a1 + k);
    a1f.q[1] = *(const uint4*)(pa + a1 + k + 16);
    b.q[0]   = *(const uint4*)(pw + k);
    b.q[1]   = *(const uint4*)(pw + k + 16);
    c0 = __builtin_amdgcn_wmma_f32_16x16x32_bf16(false, a0.v,  false, b.v, (short)0, c0, false, false);
    c1 = __builtin_amdgcn_wmma_f32_16x16x32_bf16(false, a1f.v, false, b.v, (short)0, c1, false, false);
  }
}

// 2x2 register tiling (M=32, N=32) for the bulk GEMMs.
__device__ __forceinline__ void wmma_m2n2(floatx8& c00, floatx8& c01,
                                          floatx8& c10, floatx8& c11,
    const unsigned short* __restrict__ A, size_t lda,
    const unsigned short* __restrict__ W, size_t ldw,
    int K, int lane)
{
  const unsigned short* pa = A + (size_t)(lane & 15) * lda + ((lane >> 4) << 3);
  const unsigned short* pw = W + (size_t)(lane & 15) * ldw + ((lane >> 4) << 3);
  const size_t a1 = (size_t)16 * lda;
  const size_t w1 = (size_t)16 * ldw;
  for (int k = 0; k < K; k += 32) {
    BV16 a0, a1f, b0, b1;
    a0.q[0]  = *(const uint4*)(pa + k);
    a0.q[1]  = *(const uint4*)(pa + k + 16);
    a1f.q[0] = *(const uint4*)(pa + a1 + k);
    a1f.q[1] = *(const uint4*)(pa + a1 + k + 16);
    b0.q[0]  = *(const uint4*)(pw + k);
    b0.q[1]  = *(const uint4*)(pw + k + 16);
    b1.q[0]  = *(const uint4*)(pw + w1 + k);
    b1.q[1]  = *(const uint4*)(pw + w1 + k + 16);
    c00 = __builtin_amdgcn_wmma_f32_16x16x32_bf16(false, a0.v,  false, b0.v, (short)0, c00, false, false);
    c01 = __builtin_amdgcn_wmma_f32_16x16x32_bf16(false, a0.v,  false, b1.v, (short)0, c01, false, false);
    c10 = __builtin_amdgcn_wmma_f32_16x16x32_bf16(false, a1f.v, false, b0.v, (short)0, c10, false, false);
    c11 = __builtin_amdgcn_wmma_f32_16x16x32_bf16(false, a1f.v, false, b1.v, (short)0, c11, false, false);
  }
}

// ---------------------------------------------------------------------------
// Elementwise helpers
// ---------------------------------------------------------------------------
__global__ void cvt_bf16_kernel(const float* __restrict__ src,
                                unsigned short* __restrict__ dst, int n) {
  int i = blockIdx.x * blockDim.x + threadIdx.x;
  int stride = gridDim.x * blockDim.x;
  for (; i < n; i += stride) dst[i] = f2bf(src[i]);
}

// h0: [B, L, H]  ->  hf/hb: [L, B, H]
__global__ void init_h_kernel(const float* __restrict__ h0,
                              float* __restrict__ hf,
                              unsigned short* __restrict__ hb) {
  int i = blockIdx.x * blockDim.x + threadIdx.x;
  if (i >= kL * kB * kH) return;
  int h = i % kH;
  int b = (i / kH) % kB;
  int l = i / (kH * kB);
  float v = h0[((size_t)b * kL + l) * kH + h];
  hf[i] = v;
  hb[i] = f2bf(v);
}

// hf: [L, B, H] -> out: [B, L, H]
__global__ void hid_out_kernel(const float* __restrict__ hf,
                               float* __restrict__ out) {
  int i = blockIdx.x * blockDim.x + threadIdx.x;
  if (i >= kL * kB * kH) return;
  int h = i % kH;
  int b = (i / kH) % kB;
  int l = i / (kH * kB);
  out[((size_t)b * kL + l) * kH + h] = hf[i];
}

// ---------------------------------------------------------------------------
// Bulk: layer-0 input projections for all timesteps (bf16 out).
// Each wave: one s (M=32 rows = all b of that s), one gate, one 32-col band.
// waves = 512 * 3 * 32 = 49152
// ---------------------------------------------------------------------------
__global__ void __launch_bounds__(256)
xproj_kernel(const unsigned short* __restrict__ xbf,   // [B,S,I]
             const unsigned short* __restrict__ wz,
             const unsigned short* __restrict__ wr,
             const unsigned short* __restrict__ wg,
             unsigned short* __restrict__ oz,
             unsigned short* __restrict__ orr,
             unsigned short* __restrict__ og) {
  int wave = (blockIdx.x * blockDim.x + threadIdx.x) >> 5;
  int lane = threadIdx.x & 31;
  int tn2  = wave & 31;
  int gate = (wave >> 5) % 3;
  int s    = wave / 96;
  const unsigned short* A = xbf + (size_t)s * kI;        // rows are b, lda=S*I
  const unsigned short* W = (gate == 0 ? wz : gate == 1 ? wr : wg)
                            + (size_t)tn2 * 32 * kI;
  unsigned short* out     =  gate == 0 ? oz : gate == 1 ? orr : og;
  floatx8 c00 = {}, c01 = {}, c10 = {}, c11 = {};
  wmma_m2n2(c00, c01, c10, c11, A, (size_t)kS * kI, W, kI, kI, lane);
  int col0 = tn2 * 32 + (lane & 15);
  int rb   = ((lane >> 4) << 3);
  size_t m0 = (size_t)s * 32;
  for (int v = 0; v < 8; ++v) {
    size_t r0 = (m0 + rb + v) * kH;
    size_t r1 = (m0 + 16 + rb + v) * kH;
    out[r0 + col0]      = f2bf(c00[v]);
    out[r0 + col0 + 16] = f2bf(c01[v]);
    out[r1 + col0]      = f2bf(c10[v]);
    out[r1 + col0 + 16] = f2bf(c11[v]);
  }
}

// ---------------------------------------------------------------------------
// Per-step fused z/r kernel.
// units = gate(2) x tn(64) = 128; each unit has 4 K-slices (one wave each,
// full M=32). 8 waves/block = 2 units. Grid: 64 blocks x 256 threads.
// Partials reduced through LDS; slice-0 wave finalizes.
// ---------------------------------------------------------------------------
__global__ void __launch_bounds__(256)
gru_zr_kernel(const unsigned short* __restrict__ xz,   // [32,H] bf16 or null
              const unsigned short* __restrict__ xr,
              const unsigned short* __restrict__ inp,  // [32,H] bf16 or null
              const unsigned short* __restrict__ wxz,
              const unsigned short* __restrict__ wxr,
              const unsigned short* __restrict__ hbf,  // [32,H] bf16
              const float* __restrict__ hf,            // [32,H] f32
              const unsigned short* __restrict__ whz,
              const unsigned short* __restrict__ whr,
              const float* __restrict__ bz,
              const float* __restrict__ br,
              float* __restrict__ zout,
              unsigned short* __restrict__ rhout) {
  __shared__ float red[8][32][17];
  int wid  = threadIdx.x >> 5;
  int lane = threadIdx.x & 31;
  int unit = blockIdx.x * 2 + (wid >> 2);
  int ks   = wid & 3;
  int gate = unit & 1;                   // 0 = z, 1 = r
  int tn   = unit >> 1;                  // 0..63
  const unsigned short* xp = gate ? xr : xz;
  const unsigned short* wx = gate ? wxr : wxz;
  const unsigned short* wh = gate ? whr : whz;
  const float* bias        = gate ? br  : bz;
  int k0 = ks * 256, k1 = k0 + 256;
  floatx8 c0 = {}, c1 = {};
  if (inp)
    wmma_m2(c0, c1, inp, kH, wx + (size_t)tn * 16 * kH, kH, k0, k1, lane);
  wmma_m2(c0, c1, hbf, kH, wh + (size_t)tn * 16 * kH, kH, k0, k1, lane);
  for (int v = 0; v < 8; ++v) {
    red[wid][lane][v]     = c0[v];
    red[wid][lane][v + 8] = c1[v];
  }
  __syncthreads();
  if (ks != 0) return;
  for (int j = 1; j < 4; ++j)
    for (int v = 0; v < 8; ++v) {
      c0[v] += red[wid + j][lane][v];
      c1[v] += red[wid + j][lane][v + 8];
    }
  int col = tn * 16 + (lane & 15);
  int rb  = ((lane >> 4) << 3);
  float bb = bias[col];
  for (int v = 0; v < 8; ++v) {
    size_t i0 = (size_t)(rb + v) * kH + col;
    size_t i1 = (size_t)(16 + rb + v) * kH + col;
    float p0 = c0[v] + bb, p1 = c1[v] + bb;
    if (xp) { p0 += bf2f(xp[i0]); p1 += bf2f(xp[i1]); }
    float s0 = 1.0f / (1.0f + __expf(-p0));
    float s1 = 1.0f / (1.0f + __expf(-p1));
    if (gate == 0) { zout[i0] = s0; zout[i1] = s1; }
    else { rhout[i0] = f2bf(s0 * hf[i0]); rhout[i1] = f2bf(s1 * hf[i1]); }
  }
}

// ---------------------------------------------------------------------------
// Per-step fused g / state-update kernel.
// units = tn(64); 4 K-slices each; 8 waves/block = 2 units -> 32 blocks.
// ---------------------------------------------------------------------------
__global__ void __launch_bounds__(256)
gru_g_kernel(const unsigned short* __restrict__ xg,    // [32,H] bf16 or null
             const unsigned short* __restrict__ inp,   // [32,H] bf16 or null
             const unsigned short* __restrict__ wxg,
             const unsigned short* __restrict__ rh,    // [32,H] bf16
             const unsigned short* __restrict__ whg,
             const float* __restrict__ bg,
             const float* __restrict__ zin,            // [32,H] f32
             float* __restrict__ hf,                   // in/out [32,H]
             unsigned short* __restrict__ hbf,         // out [32,H]
             unsigned short* __restrict__ hall) {      // [32,H] slice or null
  __shared__ float red[8][32][17];
  int wid  = threadIdx.x >> 5;
  int lane = threadIdx.x & 31;
  int tn   = blockIdx.x * 2 + (wid >> 2);              // 0..63
  int ks   = wid & 3;
  int k0 = ks * 256, k1 = k0 + 256;
  floatx8 c0 = {}, c1 = {};
  if (inp)
    wmma_m2(c0, c1, inp, kH, wxg + (size_t)tn * 16 * kH, kH, k0, k1, lane);
  wmma_m2(c0, c1, rh, kH, whg + (size_t)tn * 16 * kH, kH, k0, k1, lane);
  for (int v = 0; v < 8; ++v) {
    red[wid][lane][v]     = c0[v];
    red[wid][lane][v + 8] = c1[v];
  }
  __syncthreads();
  if (ks != 0) return;
  for (int j = 1; j < 4; ++j)
    for (int v = 0; v < 8; ++v) {
      c0[v] += red[wid + j][lane][v];
      c1[v] += red[wid + j][lane][v + 8];
    }
  int col = tn * 16 + (lane & 15);
  int rb  = ((lane >> 4) << 3);
  float bb = bg[col];
  for (int v = 0; v < 8; ++v) {
    size_t i0 = (size_t)(rb + v) * kH + col;
    size_t i1 = (size_t)(16 + rb + v) * kH + col;
    float p0 = c0[v] + bb, p1 = c1[v] + bb;
    if (xg) { p0 += bf2f(xg[i0]); p1 += bf2f(xg[i1]); }
    float g0 = tanhf(p0), g1 = tanhf(p1);
    float z0 = zin[i0],   z1 = zin[i1];
    float hn0 = z0 * hf[i0] + (1.0f - z0) * g0;
    float hn1 = z1 * hf[i1] + (1.0f - z1) * g1;
    hf[i0] = hn0; hf[i1] = hn1;
    unsigned short b0 = f2bf(hn0), b1 = f2bf(hn1);
    hbf[i0] = b0; hbf[i1] = b1;
    if (hall) { hall[i0] = b0; hall[i1] = b1; }
  }
}

// ---------------------------------------------------------------------------
// Bulk: out[b,s,:] = h1all[(s*B+b),:] @ Wy^T + by.  2x2 tiled.
// waves = 512(mt2) * 32(tn2) = 16384
// ---------------------------------------------------------------------------
__global__ void __launch_bounds__(256)
y_kernel(const unsigned short* __restrict__ h1all,     // [S*B, H] bf16
         const unsigned short* __restrict__ wy,        // [O, H] bf16
         const float* __restrict__ by,
         float* __restrict__ out) {                    // [B, S, O]
  int wave = (blockIdx.x * blockDim.x + threadIdx.x) >> 5;
  int lane = threadIdx.x & 31;
  int tn2  = wave & 31;
  int mt2  = wave >> 5;                  // 0..511
  floatx8 c00 = {}, c01 = {}, c10 = {}, c11 = {};
  wmma_m2n2(c00, c01, c10, c11,
            h1all + (size_t)mt2 * 32 * kH, kH,
            wy + (size_t)tn2 * 32 * kH, kH, kH, lane);
  int col0 = tn2 * 32 + (lane & 15);
  int rb   = ((lane >> 4) << 3);
  float bb0 = by[col0], bb1 = by[col0 + 16];
  for (int v = 0; v < 8; ++v) {
    int ma = mt2 * 32 + rb + v;
    int mb = ma + 16;
    size_t oa = ((size_t)(ma & 31) * kS + (ma >> 5)) * kO;
    size_t ob = ((size_t)(mb & 31) * kS + (mb >> 5)) * kO;
    out[oa + col0]      = c00[v] + bb0;
    out[oa + col0 + 16] = c01[v] + bb1;
    out[ob + col0]      = c10[v] + bb0;
    out[ob + col0 + 16] = c11[v] + bb1;
  }
}

// ---------------------------------------------------------------------------
// Host launcher
// ---------------------------------------------------------------------------
extern "C" void kernel_launch(void* const* d_in, const int* in_sizes, int n_in,
                              void* d_out, int out_size, void* d_ws, size_t ws_size,
                              hipStream_t stream) {
  (void)in_sizes; (void)n_in; (void)out_size; (void)ws_size;
  const float* x   = (const float*)d_in[0];   // [B,S,I]
  const float* h0  = (const float*)d_in[1];   // [B,L,H]
  const float* Wxz = (const float*)d_in[2];   // [L,H,I]
  const float* Wxr = (const float*)d_in[3];
  const float* Wxg = (const float*)d_in[4];
  const float* Whz = (const float*)d_in[5];   // [L,H,H]
  const float* Whr = (const float*)d_in[6];
  const float* Whg = (const float*)d_in[7];
  const float* bz  = (const float*)d_in[8];   // [L,H]
  const float* br  = (const float*)d_in[9];
  const float* bg  = (const float*)d_in[10];
  const float* Wy  = (const float*)d_in[11];  // [O,H]
  const float* by  = (const float*)d_in[12];  // [O]
  float* out = (float*)d_out;                 // [B,S,O] then [B,L,H]

  // Workspace carve-out (~195 MB total).
  char* base = (char*)d_ws;
  size_t off = 0;
  auto alloc = [&](size_t bytes) -> char* {
    char* p = base + off;
    off = (off + bytes + 255) & ~(size_t)255;
    return p;
  };
  const size_t nX  = (size_t)kB * kS * kI;
  const size_t nWx = (size_t)kL * kH * kI;
  const size_t nWh = (size_t)kL * kH * kH;
  const size_t nWy = (size_t)kO * kH;
  const size_t nSB = (size_t)kS * kB * kH;
  const size_t nBH = (size_t)kB * kH;

  unsigned short* xbf    = (unsigned short*)alloc(nX * 2);
  unsigned short* wxz_b  = (unsigned short*)alloc(nWx * 2);
  unsigned short* wxr_b  = (unsigned short*)alloc(nWx * 2);
  unsigned short* wxg_b  = (unsigned short*)alloc(nWx * 2);
  unsigned short* whz_b  = (unsigned short*)alloc(nWh * 2);
  unsigned short* whr_b  = (unsigned short*)alloc(nWh * 2);
  unsigned short* whg_b  = (unsigned short*)alloc(nWh * 2);
  unsigned short* wy_b   = (unsigned short*)alloc(nWy * 2);
  unsigned short* xz_all = (unsigned short*)alloc(nSB * 2);
  unsigned short* xr_all = (unsigned short*)alloc(nSB * 2);
  unsigned short* xg_all = (unsigned short*)alloc(nSB * 2);
  unsigned short* h1_all = (unsigned short*)alloc(nSB * 2);
  float*          hf     = (float*)alloc((size_t)kL * nBH * 4);   // [L,B,H]
  unsigned short* hbf    = (unsigned short*)alloc((size_t)kL * nBH * 2);
  float*          zbuf   = (float*)alloc(nBH * 4);
  unsigned short* rhbuf  = (unsigned short*)alloc(nBH * 2);

  // --- 1) bf16 conversions -------------------------------------------------
  auto cvt = [&](const float* s, unsigned short* d, size_t n) {
    int blocks = (int)((n + 256 * 8 - 1) / (256 * 8));
    if (blocks > 4096) blocks = 4096;
    cvt_bf16_kernel<<<blocks, 256, 0, stream>>>(s, d, (int)n);
  };
  cvt(x,   xbf,   nX);
  cvt(Wxz, wxz_b, nWx);  cvt(Wxr, wxr_b, nWx);  cvt(Wxg, wxg_b, nWx);
  cvt(Whz, whz_b, nWh);  cvt(Whr, whr_b, nWh);  cvt(Whg, whg_b, nWh);
  cvt(Wy,  wy_b,  nWy);
  init_h_kernel<<<(kL * kB * kH + 255) / 256, 256, 0, stream>>>(h0, hf, hbf);

  // --- 2) bulk precompute of layer-0 input projections --------------------
  // waves = 512 * 3 * 32 = 49152 -> 6144 blocks of 256
  xproj_kernel<<<6144, 256, 0, stream>>>(xbf, wxz_b, wxr_b, wxg_b,
                                         xz_all, xr_all, xg_all);

  // --- 3) sequential GRU scan ---------------------------------------------
  const size_t HI = (size_t)kH * kI;   // per-layer weight stride
  for (int t = 0; t < kS; ++t) {
    const size_t tOff = (size_t)t * nBH;
    // layer 0: addend = precomputed x-projection
    gru_zr_kernel<<<64, 256, 0, stream>>>(
        xz_all + tOff, xr_all + tOff, nullptr, nullptr, nullptr,
        hbf, hf, whz_b, whr_b, bz, br, zbuf, rhbuf);
    gru_g_kernel<<<32, 256, 0, stream>>>(
        xg_all + tOff, nullptr, nullptr, rhbuf, whg_b, bg,
        zbuf, hf, hbf, nullptr);
    // layer 1: input = layer-0 h_new (bf16 shadow), x-side GEMM fused in
    gru_zr_kernel<<<64, 256, 0, stream>>>(
        nullptr, nullptr, hbf, wxz_b + HI, wxr_b + HI,
        hbf + nBH, hf + nBH, whz_b + HI, whr_b + HI,
        bz + kH, br + kH, zbuf, rhbuf);
    gru_g_kernel<<<32, 256, 0, stream>>>(
        nullptr, hbf, wxg_b + HI, rhbuf, whg_b + HI, bg + kH,
        zbuf, hf + nBH, hbf + nBH, h1_all + tOff);
  }

  // --- 4) output projection + final hidden state ---------------------------
  // waves = 16384 -> 2048 blocks of 256
  y_kernel<<<2048, 256, 0, stream>>>(h1_all, wy_b, by, out);
  hid_out_kernel<<<(kL * kB * kH + 255) / 256, 256, 0, stream>>>(
      hf, out + (size_t)kB * kS * kO);
}
